// SimpleRNNClassifier_38637525795520
// MI455X (gfx1250) — matrix operand order repaired and checked by reading, not compiled, not run on hardware
//
#include <hip/hip_runtime.h>

typedef __attribute__((ext_vector_type(16))) _Float16 v16h;
typedef __attribute__((ext_vector_type(8)))  _Float16 v8h;
typedef __attribute__((ext_vector_type(2)))  _Float16 v2h;
typedef __attribute__((ext_vector_type(8)))  float    v8f;
typedef __attribute__((ext_vector_type(4)))  float    v4f;

#define EMB   16
#define HID   32
#define SEQ   512
#define NCLS  2
#define ROWS_PER_WAVE   16
#define WAVES_PER_BLOCK 4
#define ROWS_PER_BLOCK  (ROWS_PER_WAVE * WAVES_PER_BLOCK)
#define HPAD  (HID + 8)   // 80B LDS row stride: breaks bank conflicts, keeps 16B alignment

// Hardware tanh: single TRANS32 op on gfx1250.
#if __has_builtin(__builtin_amdgcn_tanhf)
__device__ __forceinline__ float hw_tanh(float x) { return __builtin_amdgcn_tanhf(x); }
#else
__device__ __forceinline__ float hw_tanh(float x) {
    float r;
    asm volatile("v_tanh_f32 %0, %1\n\tv_nop\n\tv_nop" : "=v"(r) : "v"(x));
    return r;
}
#endif

// K-permutation absorbed into the W_hh B-fragment so the pair-interleaved LDS
// image of h can be fed to WMMA exactly as loaded (no per-step shuffles).
__constant__ int PERM_LO[16] = {0,16,1,17,2,18,3,19, 8,24,9,25,10,26,11,27};
__constant__ int PERM_HI[16] = {4,20,5,21,6,22,7,23, 12,28,13,29,14,30,15,31};

__global__ __launch_bounds__(128) void rnn_fused_kernel(
    const int*   __restrict__ x,      // [B, SEQ] token ids
    const float* __restrict__ emb,    // [VOCAB, EMB]
    const float* __restrict__ W_ih,   // [HID, EMB]
    const float* __restrict__ W_hh,   // [HID, HID]
    const float* __restrict__ b_ih,   // [HID]
    const float* __restrict__ b_hh,   // [HID]
    const float* __restrict__ W_fc,   // [NCLS, HID]
    const float* __restrict__ b_fc,   // [NCLS]
    float*       __restrict__ out)    // [B, NCLS]
{
    // Wave-private hidden state, pair-interleaved halves per row.
    __shared__ _Float16 h16[WAVES_PER_BLOCK][ROWS_PER_WAVE][HPAD];
    __shared__ float    hf [WAVES_PER_BLOCK][ROWS_PER_WAVE][HID];

    const int tid  = threadIdx.x;
    const int w    = tid >> 5;
    const int lane = tid & 31;
    const int hi   = lane >> 4;
    const int nn   = lane & 15;
    const int rowBase = blockIdx.x * ROWS_PER_BLOCK + w * ROWS_PER_WAVE;

    // ---- constant B-fragments (weights) resident in registers ----
    v16h bih[2], bhh[2];
    v8f  cbias[2];
    #pragma unroll
    for (int t = 0; t < 2; ++t) {
        const int n = t * 16 + nn;
        v16h bi;
        #pragma unroll
        for (int e = 0; e < 16; ++e)
            bi[e] = (e < 8) ? (_Float16)W_ih[n * EMB + 8 * hi + e] : (_Float16)0.f;   // K=16..31 zero pad
        bih[t] = bi;
        v16h bh;
        #pragma unroll
        for (int f = 0; f < 16; ++f) {
            const int k = (hi == 0) ? PERM_LO[f] : PERM_HI[f];
            bh[f] = (_Float16)W_hh[n * HID + k];
        }
        bhh[t] = bh;
        const float bs = b_ih[n] + b_hh[n];
        #pragma unroll
        for (int i = 0; i < 8; ++i) cbias[t][i] = bs;   // persistent C operand
    }

    // Zero this wave's hidden-state slice.
    for (int i = lane; i < ROWS_PER_WAVE * HPAD; i += 32)
        (&h16[w][0][0])[i] = (_Float16)0.f;
    __builtin_amdgcn_wave_barrier();

    const int m = nn;  // A-matrix row owned by this lane
    const int* __restrict__ xrow = x + (long)(rowBase + m) * SEQ;

    // ---- 3-stage pipeline: rows for s and s+1 resident, row for s+2 in
    // flight, index for s+3 in flight. ~2 iterations to hide L2 latency. ----
    const int i0 = xrow[0];
    v4f c0 = *(const v4f*)(emb + (long)i0 * EMB + 8 * hi);
    v4f c1 = *(const v4f*)(emb + (long)i0 * EMB + 8 * hi + 4);
    const int i1 = xrow[1];
    v4f r0 = *(const v4f*)(emb + (long)i1 * EMB + 8 * hi);
    v4f r1 = *(const v4f*)(emb + (long)i1 * EMB + 8 * hi + 4);
    int idx2 = xrow[2];

    v8f acc0, acc1;

    #pragma unroll 2
    for (int s = 0; s < SEQ; ++s) {
        // Issue row load for s+2 (index already resident) and index for s+3.
        const float* nrow = emb + (long)idx2 * EMB + 8 * hi;
        v4f n0 = *(const v4f*)(nrow);
        v4f n1 = *(const v4f*)(nrow + 4);
        idx2 = xrow[(s + 3 < SEQ) ? s + 3 : SEQ - 1];
        // Speculative prefetch of the index stream (no counter, fault-safe).
        __builtin_prefetch(xrow + s + 128, 0, 0);

        // A-fragment from this step's embedding row (K=0..15 real, 16..31 zero).
        v16h a;
        #pragma unroll
        for (int e = 0; e < 4; ++e) { a[e] = (_Float16)c0[e]; a[4 + e] = (_Float16)c1[e]; }
        #pragma unroll
        for (int e = 8; e < 16; ++e) a[e] = (_Float16)0.f;

        acc0 = __builtin_amdgcn_wmma_f32_16x16x32_f16(false, a, false, bih[0], (short)0, cbias[0], false, false);
        acc1 = __builtin_amdgcn_wmma_f32_16x16x32_f16(false, a, false, bih[1], (short)0, cbias[1], false, false);

        // A-fragment: previous h (pair-interleaved LDS image, used as loaded).
        v8h hlo = *(const v8h*)&h16[w][m][8 * hi];
        v8h hhi = *(const v8h*)&h16[w][m][16 + 8 * hi];
        v16h ah;
        #pragma unroll
        for (int e = 0; e < 8; ++e) { ah[e] = hlo[e]; ah[8 + e] = hhi[e]; }

        acc0 = __builtin_amdgcn_wmma_f32_16x16x32_f16(false, ah, false, bhh[0], (short)0, acc0, false, false);
        acc1 = __builtin_amdgcn_wmma_f32_16x16x32_f16(false, ah, false, bhh[1], (short)0, acc1, false, false);

        // Hardware tanh.
        #pragma unroll
        for (int i = 0; i < 8; ++i) { acc0[i] = hw_tanh(acc0[i]); acc1[i] = hw_tanh(acc1[i]); }

        // Same-wave LDS update: packed pairs -> 2-address b32 stores.
        __builtin_amdgcn_wave_barrier();
        #pragma unroll
        for (int v = 0; v < 8; ++v) {
            const int mm = v + 8 * hi;
            v2h p;
            p[0] = (_Float16)acc0[v];
            p[1] = (_Float16)acc1[v];
            *(v2h*)&h16[w][mm][2 * nn] = p;
        }
        __builtin_amdgcn_wave_barrier();

        c0 = r0; c1 = r1; r0 = n0; r1 = n1;   // rotate pipeline
    }

    // ---- head: out = h_final @ W_fc^T + b_fc (N=2 -> VALU, f32 h) ----
    #pragma unroll
    for (int v = 0; v < 8; ++v) {
        const int mm = v + 8 * hi;
        hf[w][mm][nn]      = acc0[v];
        hf[w][mm][16 + nn] = acc1[v];
    }
    __builtin_amdgcn_wave_barrier();
    if (lane < 16) {
        const int row = rowBase + lane;
        float s0 = b_fc[0];
        float s1 = b_fc[1];
        #pragma unroll
        for (int n = 0; n < HID; ++n) {
            const float hv = hf[w][lane][n];
            s0 += hv * W_fc[n];
            s1 += hv * W_fc[HID + n];
        }
        out[row * NCLS + 0] = s0;
        out[row * NCLS + 1] = s1;
    }
}

extern "C" void kernel_launch(void* const* d_in, const int* in_sizes, int n_in,
                              void* d_out, int out_size, void* d_ws, size_t ws_size,
                              hipStream_t stream) {
    const int*   x    = (const int*)  d_in[0];
    const float* emb  = (const float*)d_in[1];
    const float* W_ih = (const float*)d_in[2];
    const float* W_hh = (const float*)d_in[3];
    const float* b_ih = (const float*)d_in[4];
    const float* b_hh = (const float*)d_in[5];
    const float* W_fc = (const float*)d_in[6];
    const float* b_fc = (const float*)d_in[7];
    float* out = (float*)d_out;

    const int B = in_sizes[0] / SEQ;                               // 4096
    const int blocks = (B + ROWS_PER_BLOCK - 1) / ROWS_PER_BLOCK;  // 64

    rnn_fused_kernel<<<blocks, WAVES_PER_BLOCK * 32, 0, stream>>>(
        x, emb, W_ih, W_hh, b_ih, b_hh, W_fc, b_fc, out);
}
